// GraphLearning_85607288143884
// MI455X (gfx1250) — matrix operand order repaired and checked by reading, not compiled, not run on hardware
//
#include <hip/hip_runtime.h>
#include <hip/hip_bf16.h>
#include <math.h>

typedef __attribute__((ext_vector_type(16))) _Float16 v16h;
typedef __attribute__((ext_vector_type(8)))  _Float16 v8h;
typedef __attribute__((ext_vector_type(4)))  _Float16 v4h;
typedef __attribute__((ext_vector_type(8)))  float    v8f;
typedef __attribute__((ext_vector_type(4)))  float    v4f;

#define NF      51
#define WIN     100
#define EMB     12
#define FIN     112      // WIN + EMB
#define HID     60
#define NPRUNE  1561
#define XQUADS  (NF * WIN / 4)   // 1275 float4 per batch sample

#define WMMA_F16(A, B, C) \
    __builtin_amdgcn_wmma_f32_16x16x32_f16(false, (A), false, (B), (short)0, (C), false, false)

// ---------------------------------------------------------------------------
// Kernel 0: serial greedy prune + GCN normalization (single wave; workgroup
// barriers for a 1-wave group are S_NOP). Writes 64x64 zero-padded norm_adj.
// ---------------------------------------------------------------------------
__global__ __launch_bounds__(32, 1)
void graph_prune_kernel(const float* __restrict__ embed, float* __restrict__ adj_out)
{
    __shared__ float A[NF * NF];
    __shared__ float emb[NF * EMB];
    __shared__ float nrm[NF];
    __shared__ float pmin[32];
    __shared__ int   pidx[32];
    __shared__ float pmax[32];
    __shared__ int   sh_i;

    const int t = threadIdx.x;

    for (int i = t; i < NF * EMB; i += 32) emb[i] = embed[i];
    __syncthreads();
    for (int i = t; i < NF; i += 32) {
        float s = 0.f;
        for (int e = 0; e < EMB; ++e) { float v = emb[i * EMB + e]; s += v * v; }
        nrm[i] = sqrtf(s);
    }
    __syncthreads();
    for (int p = t; p < NF * NF; p += 32) {
        int i = p / NF, j = p % NF;
        float d = 0.f;
        for (int e = 0; e < EMB; ++e) d += emb[i * EMB + e] * emb[j * EMB + e];
        A[p] = d / (nrm[i] * nrm[j]);
    }
    if (t == 0) sh_i = 0;
    __syncthreads();

    for (int iter = 0; iter < 200000; ++iter) {
        if (sh_i >= NPRUNE) break;
        float mv = 3.4e38f; int mi = 0; float xv = -3.4e38f;
        for (int p = t; p < NF * NF; p += 32) {
            float v = A[p];
            if (v < mv) { mv = v; mi = p; }
            if (v > xv) xv = v;
        }
        pmin[t] = mv; pidx[t] = mi; pmax[t] = xv;
        __syncthreads();
        if (t == 0) {
            float bmv = pmin[0]; int bmi = pidx[0]; float bxv = pmax[0];
            for (int l = 1; l < 32; ++l) {
                if (pmin[l] < bmv || (pmin[l] == bmv && pidx[l] < bmi)) { bmv = pmin[l]; bmi = pidx[l]; }
                if (pmax[l] > bxv) bxv = pmax[l];
            }
            int r = bmi / NF, c = bmi % NF;
            float v = A[bmi];
            float rs = 0.f, cs = 0.f;
            for (int j = 0; j < NF; ++j) rs += A[r * NF + j];
            for (int j = 0; j < NF; ++j) cs += A[j * NF + c];
            bool ok = ((rs - v) > 0.f) || ((cs - v) > 0.f);
            A[bmi] = ok ? 0.f : bxv;
            if (ok) sh_i = sh_i + 1;
        }
        __syncthreads();
    }

    __syncthreads();
    for (int i = t; i < NF; i += 32) {
        float d = 0.f;
        for (int j = 0; j < NF; ++j)
            d += ((A[i * NF + j] != 0.f) ? 1.f : 0.f) + ((i == j) ? 1.f : 0.f);
        nrm[i] = 1.f / sqrtf(d);
    }
    __syncthreads();
    for (int p = t; p < 64 * 64; p += 32) {
        int i = p >> 6, j = p & 63;
        float val = 0.f;
        if (i < NF && j < NF) {
            float ah = ((A[i * NF + j] != 0.f) ? 1.f : 0.f) + ((i == j) ? 1.f : 0.f);
            val = nrm[i] * ah * nrm[j];
        }
        adj_out[p] = val;
    }
}

// ---------------------------------------------------------------------------
// Packed-fragment address helpers (wave32 WMMA 16-bit layouts).
// ---------------------------------------------------------------------------
__device__ __forceinline__ int a_pack_addr(int m, int k, int KT)
{
    int mt = m >> 4, kt = k >> 5, r = k & 31;
    int lane = (m & 15) + ((r >> 4) << 4);
    int e = r & 15;
    return (((mt * KT + kt) * 32 + lane) << 4) + e;
}

// ---------------------------------------------------------------------------
// Fused GCN + MLP. All GEMM operands live in LDS in fragment order (one 32B
// vector per lane per fragment). x is double-pumped: async global->LDS loads
// for batch b+1 (ASYNCcnt) overlap the WMMA work of batch b.
// ---------------------------------------------------------------------------
__global__ __launch_bounds__(128, 1)
void fused_gcn_mlp_kernel(const float* __restrict__ x,    const float* __restrict__ embed,
                          const float* __restrict__ gcn_w, const float* __restrict__ gcn_b,
                          const float* __restrict__ w1,    const float* __restrict__ b1,
                          const float* __restrict__ w2,    const float* __restrict__ b2,
                          const float* __restrict__ adjw,  float* __restrict__ out, int batch)
{
    __shared__ v16h xcatp[4 * 4 * 32];   // A for GEMM1 (64 x 128)          16 KB
    __shared__ v16h h1p  [2 * 7 * 32];   // B for GEMM2 (64 x 112)          14 KB
    __shared__ v16h h2p  [4 * 4 * 32];   // A for GEMM3 (64 x 128)          16 KB
    __shared__ v16h h3p  [4 * 2 * 32];   // A for GEMM4 (64 x 64)            8 KB
    __shared__ v16h adjp [4 * 2 * 32];   // A for GEMM2 (64 x 64)            8 KB
    __shared__ v16h gwTp [4 * 7 * 32];   // B gcn_w^T  (128 x 112)          28 KB
    __shared__ v16h w1Tp [4 * 4 * 32];   // B w1^T     (128 x 64)           16 KB
    __shared__ v16h w2Tp [2 * 7 * 32];   // B w2^T     (64 x 112)           14 KB
    __shared__ __align__(16) float xraw[NF * WIN];   // async staging        20 KB
    __shared__ float bg[112], bh[64], bo[112];

    _Float16* xcatf = (_Float16*)xcatp;
    _Float16* h1f   = (_Float16*)h1p;
    _Float16* h2f   = (_Float16*)h2p;
    _Float16* h3f   = (_Float16*)h3p;

    const int t    = threadIdx.x;
    const int lane = t & 31;
    const int wv   = t >> 5;

    // ---- one-time init ----
    for (int i = t; i < 4 * 4 * 32 * 16; i += 128) xcatf[i] = (_Float16)0.f;   // pads load-bearing
    for (int i = t; i < 4 * 4 * 32 * 16; i += 128) h2f[i]   = (_Float16)0.f;   // cols 112..127 stay 0

    for (int i = t; i < 4 * 7 * 32 * 16; i += 128) {            // gcn_w^T packed-B
        int f = i >> 9, l = (i >> 4) & 31, e = i & 15;
        int kt = f / 7, nt = f % 7;
        int k = kt * 32 + ((l >= 16) ? 16 : 0) + e;
        int n = nt * 16 + (l & 15);
        ((_Float16*)gwTp)[i] = (_Float16)((k < FIN && n < WIN) ? gcn_w[n * FIN + k] : 0.f);
    }
    for (int i = t; i < 4 * 4 * 32 * 16; i += 128) {            // w1^T packed-B
        int f = i >> 9, l = (i >> 4) & 31, e = i & 15;
        int kt = f >> 2, nt = f & 3;
        int k = kt * 32 + ((l >= 16) ? 16 : 0) + e;
        int n = nt * 16 + (l & 15);
        ((_Float16*)w1Tp)[i] = (_Float16)((k < WIN && n < HID) ? w1[n * WIN + k] : 0.f);
    }
    for (int i = t; i < 2 * 7 * 32 * 16; i += 128) {            // w2^T packed-B
        int f = i >> 9, l = (i >> 4) & 31, e = i & 15;
        int kt = f / 7, nt = f % 7;
        int k = kt * 32 + ((l >= 16) ? 16 : 0) + e;
        int n = nt * 16 + (l & 15);
        ((_Float16*)w2Tp)[i] = (_Float16)((k < HID && n < WIN) ? w2[n * HID + k] : 0.f);
    }
    for (int i = t; i < 4 * 2 * 32 * 16; i += 128) {            // norm_adj packed-A
        int f = i >> 9, l = (i >> 4) & 31, e = i & 15;
        int mt = f >> 1, kt = f & 1;
        int m = mt * 16 + (l & 15);
        int k = kt * 32 + ((l >= 16) ? 16 : 0) + e;
        ((_Float16*)adjp)[i] = (_Float16)adjw[(m << 6) + k];
    }
    for (int i = t; i < 112; i += 128) bg[i] = (i < WIN) ? gcn_b[i] : 0.f;
    for (int i = t; i < 64;  i += 128) bh[i] = (i < HID) ? b1[i] : 0.f;
    for (int i = t; i < 112; i += 128) bo[i] = (i < WIN) ? b2[i] : 0.f;
    __syncthreads();
    for (int i = t; i < NF * EMB; i += 128) {                   // embed cols: batch-invariant
        int m = i / EMB, e = i % EMB;
        xcatf[a_pack_addr(m, WIN + e, 4)] = (_Float16)embed[i];
    }
    __syncthreads();

    const v16h a2_0 = adjp[(wv * 2 + 0) * 32 + lane];
    const v16h a2_1 = adjp[(wv * 2 + 1) * 32 + lane];

    const int crow = (lane < 16) ? 0 : 8;
    const int ncol = lane & 15;
    const int m0    = wv * 16 + crow;          // h1 D -> packed-B store coords
    const int kt2   = m0 >> 5;
    const int r0    = m0 & 31;
    const int lane2 = ncol + ((r0 < 16) ? 0 : 16);
    const int e2    = r0 & 15;

    const unsigned xraw_lds = (unsigned)(uintptr_t)(void*)&xraw[0];

    // prologue: async-stage the first batch
    {
        const float* xb = x + (size_t)blockIdx.x * (NF * WIN);
        for (int q = t; q < XQUADS; q += 128) {
            unsigned lo = xraw_lds + (unsigned)(q * 16);
            asm volatile("global_load_async_to_lds_b128 %0, %1, %2 offset:0"
                         :: "v"(lo), "v"(q * 16), "s"(xb) : "memory");
        }
    }

    for (int b = blockIdx.x; b < batch; b += gridDim.x) {
        // wait for this batch's staged x, convert f32 -> f16 packed-A fragments
        asm volatile("s_wait_asynccnt 0x0" ::: "memory");
        for (int q = t; q < XQUADS; q += 128) {
            int m  = q / 25;                 // 25 quads per 100-wide row
            int w4 = (q % 25) * 4;           // quad never crosses an 8-elem sub-block
            v4f vv = *(const v4f*)&xraw[q * 4];
            v4h hv;
            hv[0] = (_Float16)vv.x; hv[1] = (_Float16)vv.y;
            hv[2] = (_Float16)vv.z; hv[3] = (_Float16)vv.w;
            *(v4h*)(xcatf + a_pack_addr(m, w4, 4)) = hv;
        }
        __syncthreads();

        // overlap next batch's x stream with this batch's WMMA work
        int bn = b + gridDim.x;
        if (bn < batch) {
            const float* xb = x + (size_t)bn * (NF * WIN);
            for (int q = t; q < XQUADS; q += 128) {
                unsigned lo = xraw_lds + (unsigned)(q * 16);
                asm volatile("global_load_async_to_lds_b128 %0, %1, %2 offset:0"
                             :: "v"(lo), "v"(q * 16), "s"(xb) : "memory");
            }
        }

        // ---- GEMM1: h1 = [x|emb] @ gcn_w^T   (M64 K128 N112) ----
        {
            v16h a0 = xcatp[(wv * 4 + 0) * 32 + lane];
            v16h a1 = xcatp[(wv * 4 + 1) * 32 + lane];
            v16h a2 = xcatp[(wv * 4 + 2) * 32 + lane];
            v16h a3 = xcatp[(wv * 4 + 3) * 32 + lane];
            for (int nt = 0; nt < 7; ++nt) {
                v8f acc = {};
                acc = WMMA_F16(a0, gwTp[(0 * 7 + nt) * 32 + lane], acc);
                acc = WMMA_F16(a1, gwTp[(1 * 7 + nt) * 32 + lane], acc);
                acc = WMMA_F16(a2, gwTp[(2 * 7 + nt) * 32 + lane], acc);
                acc = WMMA_F16(a3, gwTp[(3 * 7 + nt) * 32 + lane], acc);
                v8h sv;
#pragma unroll
                for (int g = 0; g < 8; ++g) sv[g] = (_Float16)acc[g];
                *(v8h*)(h1f + ((((kt2 * 7 + nt) * 32 + lane2) << 4) + e2)) = sv;
            }
        }
        __syncthreads();

        // ---- GEMM2: h2 = relu(norm_adj @ h1 + gcn_b)   (M64 K64 N112) ----
        for (int nt = 0; nt < 7; ++nt) {
            v8f acc = {};
            acc = WMMA_F16(a2_0, h1p[(0 * 7 + nt) * 32 + lane], acc);
            acc = WMMA_F16(a2_1, h1p[(1 * 7 + nt) * 32 + lane], acc);
            int n = nt * 16 + ncol;
            float bias = bg[n];
#pragma unroll
            for (int g = 0; g < 8; ++g)
                h2f[a_pack_addr(wv * 16 + g + crow, n, 4)] =
                    (_Float16)fmaxf(acc[g] + bias, 0.f);
        }
        __syncthreads();

        // ---- GEMM3: h3 = tanh(h2 @ w1^T + b1)   (M64 K128 N64) ----
        {
            v16h a0 = h2p[(wv * 4 + 0) * 32 + lane];
            v16h a1 = h2p[(wv * 4 + 1) * 32 + lane];
            v16h a2 = h2p[(wv * 4 + 2) * 32 + lane];
            v16h a3 = h2p[(wv * 4 + 3) * 32 + lane];
            for (int nt = 0; nt < 4; ++nt) {
                v8f acc = {};
                acc = WMMA_F16(a0, w1Tp[(0 * 4 + nt) * 32 + lane], acc);
                acc = WMMA_F16(a1, w1Tp[(1 * 4 + nt) * 32 + lane], acc);
                acc = WMMA_F16(a2, w1Tp[(2 * 4 + nt) * 32 + lane], acc);
                acc = WMMA_F16(a3, w1Tp[(3 * 4 + nt) * 32 + lane], acc);
                int n = nt * 16 + ncol;
                float bias = bh[n];
#pragma unroll
                for (int g = 0; g < 8; ++g)
                    h3f[a_pack_addr(wv * 16 + g + crow, n, 2)] =
                        (_Float16)tanhf(acc[g] + bias);
            }
        }
        __syncthreads();

        // ---- GEMM4: out = tanh(h3 @ w2^T + b2)   (M64 K64 N112) ----
        {
            v16h a0 = h3p[(wv * 2 + 0) * 32 + lane];
            v16h a1 = h3p[(wv * 2 + 1) * 32 + lane];
            for (int nt = 0; nt < 7; ++nt) {
                v8f acc = {};
                acc = WMMA_F16(a0, w2Tp[(0 * 7 + nt) * 32 + lane], acc);
                acc = WMMA_F16(a1, w2Tp[(1 * 7 + nt) * 32 + lane], acc);
                int n = nt * 16 + ncol;
                float bias = bo[n];
#pragma unroll
                for (int g = 0; g < 8; ++g) {
                    int m = wv * 16 + g + crow;
                    if (m < NF && n < WIN)
                        out[((size_t)(b * NF + m)) * WIN + n] = tanhf(acc[g] + bias);
                }
            }
        }
        __syncthreads();
    }
}

// ---------------------------------------------------------------------------
extern "C" void kernel_launch(void* const* d_in, const int* in_sizes, int n_in,
                              void* d_out, int out_size, void* d_ws, size_t ws_size,
                              hipStream_t stream)
{
    const float* x     = (const float*)d_in[0];
    const float* embed = (const float*)d_in[2];
    const float* gcn_w = (const float*)d_in[3];
    const float* gcn_b = (const float*)d_in[4];
    const float* w1    = (const float*)d_in[5];
    const float* b1    = (const float*)d_in[6];
    const float* w2    = (const float*)d_in[7];
    const float* b2    = (const float*)d_in[8];
    float* out   = (float*)d_out;
    float* adjws = (float*)d_ws;            // 64*64 f32 = 16 KB

    const int batch = in_sizes[0] / (NF * WIN);

    graph_prune_kernel<<<1, 32, 0, stream>>>(embed, adjws);

    int grid = (batch < 2048) ? batch : 2048;
    fused_gcn_mlp_kernel<<<grid, 128, 0, stream>>>(x, embed, gcn_w, gcn_b,
                                                   w1, b1, w2, b2, adjws, out, batch);
}
